// MixtureOfMixers_66391604462084
// MI455X (gfx1250) — matrix operand order repaired and checked by reading, not compiled, not run on hardware
//
#include <hip/hip_runtime.h>
#include <hip/hip_bf16.h>
#include <math.h>

// ---------------------------------------------------------------------------
// MixtureOfMixers on MI455X (gfx1250): top-2 sparse execution, bf16 WMMA path,
// async global->LDS staging of A tiles, 4-wide accumulator blocking per wave.
// ---------------------------------------------------------------------------

typedef __attribute__((ext_vector_type(16))) __bf16 v16bf;
typedef __attribute__((ext_vector_type(8)))  __bf16 v8bf;
typedef __attribute__((ext_vector_type(8)))  float  v8f;
typedef int vsi4 __attribute__((vector_size(16)));   // b128 payload type

#define B_    32
#define N_    1024
#define D_    768
#define E_TOK 4
#define E_CH  4
#define E_    8
#define HN_   4096
#define HD_   3072

// ---- helpers ---------------------------------------------------------------

static __device__ __forceinline__ unsigned short f2bf(float f) {
  unsigned int u = __builtin_bit_cast(unsigned int, f);
  u += 0x7FFFu + ((u >> 16) & 1u);         // round-to-nearest-even
  return (unsigned short)(u >> 16);
}

// Branchless tanh-gelu: tanh(u) = sign(u) * (1 - e^{-2|u|}) / (1 + e^{-2|u|})
// using v_exp_f32 (exp2) and v_rcp_f32 directly -> no EXEC-mask branches.
static __device__ __forceinline__ float gelu_tanh(float x) {
  float u = 0.7978845608028654f * (x + 0.044715f * x * x * x);
  float au = fabsf(u);
  float en = __builtin_amdgcn_exp2f(au * -2.8853900817779268f);  // e^{-2|u|}
  float t = (1.0f - en) * __builtin_amdgcn_rcpf(1.0f + en);
  t = copysignf(t, u);
  return 0.5f * x * (1.0f + t);
}

static __device__ __forceinline__ v16bf frag_join(v8bf lo, v8bf hi) {
  return __builtin_shufflevector(lo, hi, 0,1,2,3,4,5,6,7,8,9,10,11,12,13,14,15);
}

// A-matrix fragment (16x32 bf16): lane holds row M=lane%16; K chunks at
// k0 = kk*32 + (lane>=16?8:0) + {0..7} and the same +16.  Two b128 loads.
static __device__ __forceinline__ v16bf load_a_frag(const unsigned short* p) {
  v8bf lo = *reinterpret_cast<const v8bf*>(p);
  v8bf hi = *reinterpret_cast<const v8bf*>(p + 16);
  return frag_join(lo, hi);
}

// B-matrix fragment (32x16 bf16): lane holds col N=lane%16; K chunks at
// kk*32 + (lane>=16?8:0) + {0..7} and the same +16.  Two b128 loads.
static __device__ __forceinline__ v16bf load_b_frag(const unsigned short* p) {
  v8bf lo = *reinterpret_cast<const v8bf*>(p);
  v8bf hi = *reinterpret_cast<const v8bf*>(p + 16);
  return frag_join(lo, hi);
}

static __device__ __forceinline__ v8f wmma_bf16(v16bf a, v16bf b, v8f c) {
  return __builtin_amdgcn_wmma_f32_16x16x32_bf16(false, a, false, b,
                                                 (short)0, c, false, false);
}

// 16-byte global -> LDS copy via the CDNA5 async path (ASYNCcnt-tracked).
static __device__ __forceinline__ void cp16_g2l(unsigned short* l,
                                                const unsigned short* g) {
#if __has_builtin(__builtin_amdgcn_global_load_async_to_lds_b128)
  __builtin_amdgcn_global_load_async_to_lds_b128(
      (__attribute__((address_space(1))) vsi4*)g,
      (__attribute__((address_space(3))) vsi4*)l, 0, 0);
#else
  *reinterpret_cast<v8bf*>(l) = *reinterpret_cast<const v8bf*>(g);
#endif
}

static __device__ __forceinline__ void wait_async() {
  asm volatile("s_wait_asynccnt 0x0" ::: "memory");
}

// One A fragment feeds TW consecutive column tiles (TW accumulators).
template <int TW, int KITER>
static __device__ __forceinline__ void mm_rowtile(
    const unsigned short* aBase, const unsigned short* bBase,
    size_t bColStride, int hiHalf, v8f* c) {
#pragma unroll 2
  for (int kk = 0; kk < KITER; ++kk) {
    v16bf a = load_a_frag(aBase + kk * 32 + hiHalf * 8);
    int kb = kk * 32 + hiHalf * 8;
#pragma unroll
    for (int j = 0; j < TW; ++j)
      c[j] = wmma_bf16(a, load_b_frag(bBase + (size_t)j * bColStride + kb), c[j]);
  }
}

// ---- small kernels ---------------------------------------------------------

__global__ __launch_bounds__(256) void zero_kernel(float* out, size_t n) {
  size_t i = (size_t)blockIdx.x * blockDim.x + threadIdx.x;
  size_t stride = (size_t)gridDim.x * blockDim.x;
  for (; i < n; i += stride) out[i] = 0.0f;
}

__global__ __launch_bounds__(256) void cvt_w_kernel(const float* __restrict__ w,
                                                    unsigned short* __restrict__ o,
                                                    int n) {
  int i = blockIdx.x * blockDim.x + threadIdx.x;
  int stride = gridDim.x * blockDim.x;
  for (; i < n; i += stride) o[i] = f2bf(w[i]);
}

// x (B,N,D) fp32 -> xbf (B,N,D) bf16 and xtbf (B,D,N) bf16
__global__ __launch_bounds__(256) void cvt_x_kernel(const float* __restrict__ x,
                                                    unsigned short* __restrict__ xbf,
                                                    unsigned short* __restrict__ xtbf) {
  size_t total = (size_t)B_ * N_ * D_;
  size_t i = (size_t)blockIdx.x * blockDim.x + threadIdx.x;
  size_t stride = (size_t)gridDim.x * blockDim.x;
  for (; i < total; i += stride) {
    int d = (int)(i % D_);
    int n = (int)((i / D_) % N_);
    int b = (int)(i / ((size_t)D_ * N_));
    unsigned short v = f2bf(x[i]);
    xbf[i] = v;
    xtbf[((size_t)b * D_ + d) * N_ + n] = v;
  }
}

// One block per batch element: mean over N, 8 logits, softmax, top-2, combine.
__global__ __launch_bounds__(256) void router_kernel(const float* __restrict__ x,
                                                     const float* __restrict__ rw,
                                                     float* __restrict__ combine,
                                                     float* __restrict__ probs,
                                                     int* __restrict__ top1) {
  int b = blockIdx.x;
  int tid = threadIdx.x;
  float acc[E_];
#pragma unroll
  for (int e = 0; e < E_; ++e) acc[e] = 0.0f;
  for (int dd = 0; dd < D_ / 256; ++dd) {
    int d = tid + dd * 256;
    float s = 0.0f;
    for (int n = 0; n < N_; ++n) s += x[((size_t)b * N_ + n) * D_ + d];
    s *= (1.0f / (float)N_);
#pragma unroll
    for (int e = 0; e < E_; ++e) acc[e] += s * rw[e * D_ + d];
  }
  __shared__ float slog[E_];
  if (tid < E_) slog[tid] = 0.0f;
  __syncthreads();
#pragma unroll
  for (int e = 0; e < E_; ++e) atomicAdd(&slog[e], acc[e]);
  __syncthreads();
  if (tid == 0) {
    float l[E_];
    float m = -1e30f;
#pragma unroll
    for (int e = 0; e < E_; ++e) { l[e] = slog[e]; m = fmaxf(m, l[e]); }
    float s = 0.0f;
#pragma unroll
    for (int e = 0; e < E_; ++e) { l[e] = __expf(l[e] - m); s += l[e]; }
    float inv = 1.0f / s;
#pragma unroll
    for (int e = 0; e < E_; ++e) { l[e] *= inv; probs[b * E_ + e] = l[e]; }
    int i0 = 0;
#pragma unroll
    for (int e = 1; e < E_; ++e) if (l[e] > l[i0]) i0 = e;
    int i1 = (i0 == 0) ? 1 : 0;
#pragma unroll
    for (int e = 0; e < E_; ++e) if (e != i0 && l[e] > l[i1]) i1 = e;
    float w0 = l[i0], w1 = l[i1];
    float invw = 1.0f / (w0 + w1);
#pragma unroll
    for (int e = 0; e < E_; ++e) combine[b * E_ + e] = 0.0f;
    combine[b * E_ + i0] = w0 * invw;
    combine[b * E_ + i1] = w1 * invw;
    top1[b] = i0;
  }
}

__global__ void aux_kernel(const float* __restrict__ probs,
                           const int* __restrict__ top1,
                           float* __restrict__ out_aux) {
  __shared__ float part[E_];
  int e = threadIdx.x;
  if (e < E_) {
    float pm = 0.0f, fm = 0.0f;
    for (int b = 0; b < B_; ++b) {
      pm += probs[b * E_ + e];
      fm += (top1[b] == e) ? 1.0f : 0.0f;
    }
    part[e] = (pm / (float)B_) * (fm / (float)B_);
  }
  __syncthreads();
  if (threadIdx.x == 0) {
    float s = 0.0f;
#pragma unroll
    for (int i = 0; i < E_; ++i) s += part[i];
    *out_aux = (float)E_ * s;
  }
}

// ---- fused token-mixing expert MLP -----------------------------------------
// Block = (b, e, dtile): async-stage A (16 x N, contiguous 32KB) to LDS.
// GEMM1: (16,N)x(N,HN) -> gelu -> LDS (16x4096 bf16).  GEMM2: (16,HN)x(HN,N)
// -> weighted atomic add into out[b][n][d].
__global__ __launch_bounds__(256) void tok_kernel(const unsigned short* __restrict__ xt,
                                                  const unsigned short* __restrict__ w1,
                                                  const unsigned short* __restrict__ w2,
                                                  const float* __restrict__ b1,
                                                  const float* __restrict__ b2,
                                                  const float* __restrict__ combine,
                                                  float* __restrict__ out) {
  int blk = blockIdx.x;
  int dtile = blk % (D_ / 16);
  int tmp = blk / (D_ / 16);
  int e = tmp % E_TOK;
  int b = tmp / E_TOK;
  float wgt = combine[b * E_ + e];
  if (wgt == 0.0f) return;                       // unselected expert: retire

  extern __shared__ unsigned short smem[];
  unsigned short* aTile = smem;                  // 16 x N_   (32KB)
  unsigned short* hbuf  = smem + 16 * N_;        // 16 x HN_  (128KB)

  int tid = threadIdx.x;
  int wave = tid >> 5;
  int lane = tid & 31;
  int lo16 = lane & 15;
  int hiHalf = (lane >> 4) & 1;

  // ---- stage A tile (contiguous block of 16 rows) ----
  {
    const unsigned short* gsrc = xt + ((size_t)b * D_ + dtile * 16) * N_;
#pragma unroll
    for (int p = 0; p < (16 * N_) / (256 * 8); ++p) {   // 8 passes of b128
      int off = (tid + p * 256) * 8;
      cp16_g2l(aTile + off, gsrc + off);
    }
    wait_async();
    __syncthreads();
  }

  // ---- GEMM1: h = gelu(A @ w1^T + b1) ----
  const unsigned short* aBase1 = aTile + (size_t)lo16 * N_;
  for (int g = 0; g < HN_ / (16 * 32); ++g) {          // 8 groups of 4 tiles
    int col0 = (g * 32 + wave * 4) * 16 + lo16;
    const unsigned short* bBase = w1 + ((size_t)e * HN_ + col0) * N_;
    v8f c[4] = {};
    mm_rowtile<4, N_ / 32>(aBase1, bBase, (size_t)16 * N_, hiHalf, c);
#pragma unroll
    for (int j = 0; j < 4; ++j) {
      int col = col0 + j * 16;
      float bias = b1[e * HN_ + col];
#pragma unroll
      for (int r = 0; r < 8; ++r)
        hbuf[(size_t)(r + hiHalf * 8) * HN_ + col] = f2bf(gelu_tanh(c[j][r] + bias));
    }
  }
  __syncthreads();

  // ---- GEMM2: out = h @ w2^T + b2 (transposed store, weighted accumulate) ----
  const unsigned short* aBase2 = hbuf + (size_t)lo16 * HN_;
  for (int g = 0; g < N_ / (16 * 32); ++g) {           // 2 groups of 4 tiles
    int col0 = (g * 32 + wave * 4) * 16 + lo16;        // n
    const unsigned short* bBase = w2 + ((size_t)e * N_ + col0) * HN_;
    v8f c[4] = {};
    mm_rowtile<4, HN_ / 32>(aBase2, bBase, (size_t)16 * HN_, hiHalf, c);
#pragma unroll
    for (int j = 0; j < 4; ++j) {
      int col = col0 + j * 16;
      float bias = b2[e * N_ + col];
#pragma unroll
      for (int r = 0; r < 8; ++r) {
        int d = dtile * 16 + r + hiHalf * 8;
        atomicAdd(&out[((size_t)b * N_ + col) * D_ + d], (c[j][r] + bias) * wgt);
      }
    }
  }
}

// ---- fused channel-mixing expert MLP ---------------------------------------
__global__ __launch_bounds__(256) void ch_kernel(const unsigned short* __restrict__ xb,
                                                 const unsigned short* __restrict__ w1,
                                                 const unsigned short* __restrict__ w2,
                                                 const float* __restrict__ b1,
                                                 const float* __restrict__ b2,
                                                 const float* __restrict__ combine,
                                                 float* __restrict__ out) {
  int blk = blockIdx.x;
  int ntile = blk % (N_ / 16);
  int tmp = blk / (N_ / 16);
  int e = tmp % E_CH;
  int b = tmp / E_CH;
  float wgt = combine[b * E_ + E_TOK + e];
  if (wgt == 0.0f) return;

  extern __shared__ unsigned short smem2[];
  unsigned short* aTile = smem2;                 // 16 x D_   (24KB)
  unsigned short* gbuf  = smem2 + 16 * D_;       // 16 x HD_  (96KB)

  int tid = threadIdx.x;
  int wave = tid >> 5;
  int lane = tid & 31;
  int lo16 = lane & 15;
  int hiHalf = (lane >> 4) & 1;

  // ---- stage A tile ----
  {
    const unsigned short* gsrc = xb + ((size_t)b * N_ + ntile * 16) * D_;
#pragma unroll
    for (int p = 0; p < (16 * D_) / (256 * 8); ++p) {   // 6 passes
      int off = (tid + p * 256) * 8;
      cp16_g2l(aTile + off, gsrc + off);
    }
    wait_async();
    __syncthreads();
  }

  // ---- GEMM1: g = gelu(A @ w1^T + b1) ----
  const unsigned short* aBase1 = aTile + (size_t)lo16 * D_;
  for (int g = 0; g < HD_ / (16 * 32); ++g) {          // 6 groups of 4 tiles
    int col0 = (g * 32 + wave * 4) * 16 + lo16;
    const unsigned short* bBase = w1 + ((size_t)e * HD_ + col0) * D_;
    v8f c[4] = {};
    mm_rowtile<4, D_ / 32>(aBase1, bBase, (size_t)16 * D_, hiHalf, c);
#pragma unroll
    for (int j = 0; j < 4; ++j) {
      int col = col0 + j * 16;
      float bias = b1[e * HD_ + col];
#pragma unroll
      for (int r = 0; r < 8; ++r)
        gbuf[(size_t)(r + hiHalf * 8) * HD_ + col] = f2bf(gelu_tanh(c[j][r] + bias));
    }
  }
  __syncthreads();

  // ---- GEMM2: out = g @ w2^T + b2 ----
  const unsigned short* aBase2 = gbuf + (size_t)lo16 * HD_;
  for (int g = 0; g < 3; ++g) {                        // 3 groups of 2 tiles
    int col0 = (g * 16 + wave * 2) * 16 + lo16;        // d
    const unsigned short* bBase = w2 + ((size_t)e * D_ + col0) * HD_;
    v8f c[2] = {};
    mm_rowtile<2, HD_ / 32>(aBase2, bBase, (size_t)16 * HD_, hiHalf, c);
#pragma unroll
    for (int j = 0; j < 2; ++j) {
      int col = col0 + j * 16;
      float bias = b2[e * D_ + col];
#pragma unroll
      for (int r = 0; r < 8; ++r) {
        int n = ntile * 16 + r + hiHalf * 8;
        atomicAdd(&out[((size_t)b * N_ + n) * D_ + col], (c[j][r] + bias) * wgt);
      }
    }
  }
}

// ---- launch ----------------------------------------------------------------

extern "C" void kernel_launch(void* const* d_in, const int* in_sizes, int n_in,
                              void* d_out, int out_size, void* d_ws, size_t ws_size,
                              hipStream_t stream) {
  (void)in_sizes; (void)n_in; (void)out_size; (void)ws_size;

  const float* x   = (const float*)d_in[0];
  const float* rw  = (const float*)d_in[1];
  const float* tw1 = (const float*)d_in[2];
  const float* tb1 = (const float*)d_in[3];
  const float* tw2 = (const float*)d_in[4];
  const float* tb2 = (const float*)d_in[5];
  const float* cw1 = (const float*)d_in[6];
  const float* cb1 = (const float*)d_in[7];
  const float* cw2 = (const float*)d_in[8];
  const float* cb2 = (const float*)d_in[9];
  float* out = (float*)d_out;

  char* ws = (char*)d_ws;
  float* combine = (float*)(ws + 0);
  float* probs   = (float*)(ws + 1024);
  int*   top1    = (int*)(ws + 2048);
  size_t xelems  = (size_t)B_ * N_ * D_;
  unsigned short* xbf  = (unsigned short*)(ws + 4096);
  unsigned short* xtbf = xbf  + xelems;
  unsigned short* tw1b = xtbf + xelems;
  unsigned short* tw2b = tw1b + (size_t)E_TOK * HN_ * N_;
  unsigned short* cw1b = tw2b + (size_t)E_TOK * HN_ * N_;
  unsigned short* cw2b = cw1b + (size_t)E_CH * HD_ * D_;

  zero_kernel<<<2048, 256, 0, stream>>>(out, xelems);
  cvt_x_kernel<<<4096, 256, 0, stream>>>(x, xbf, xtbf);
  cvt_w_kernel<<<2048, 256, 0, stream>>>(tw1, tw1b, E_TOK * HN_ * N_);
  cvt_w_kernel<<<2048, 256, 0, stream>>>(tw2, tw2b, E_TOK * HN_ * N_);
  cvt_w_kernel<<<2048, 256, 0, stream>>>(cw1, cw1b, E_CH * HD_ * D_);
  cvt_w_kernel<<<2048, 256, 0, stream>>>(cw2, cw2b, E_CH * HD_ * D_);
  router_kernel<<<B_, 256, 0, stream>>>(x, rw, combine, probs, top1);
  aux_kernel<<<1, 32, 0, stream>>>(probs, top1, out + xelems);

  tok_kernel<<<B_ * E_TOK * (D_ / 16), 256, 16 * (N_ + HN_) * 2, stream>>>(
      xtbf, tw1b, tw2b, tb1, tb2, combine, out);
  ch_kernel<<<B_ * E_CH * (N_ / 16), 256, 16 * (D_ + HD_) * 2, stream>>>(
      xbf, cw1b, cw2b, cb1, cb2, combine, out);
}
